// MambaLayer_4801773437301
// MI455X (gfx1250) — matrix operand order repaired and checked by reading, not compiled, not run on hardware
//
#include <hip/hip_runtime.h>
#include <hip/hip_bf16.h>

// ---------------------------------------------------------------------------
// Mamba block forward for MI455X (gfx1250, wave32).
//   GEMMs: bf16 WMMA (v_wmma_f32_16x16x32_bf16), fp32 accumulate.
//          wave tile 32x64 (8 accumulators), block tile 128x128, 8 waves.
//   Scan:  fp32, 1 thread per (b,d) channel, h[16] in VGPRs, B/C tiles in LDS.
// ---------------------------------------------------------------------------

#define B_   2
#define L_   2048
#define DM_  1024
#define DI_  2048
#define DS_  16
#define DC_  4
#define DTR_ 64
#define M_   (B_ * L_)          // 4096 tokens

typedef __bf16 bf16_t;
typedef __attribute__((ext_vector_type(8)))  float  v8f;
typedef __attribute__((ext_vector_type(8)))  bf16_t v8bf;
typedef __attribute__((ext_vector_type(16))) bf16_t v16bf;

__device__ __forceinline__ bf16_t f2bf(float f) {
    unsigned u = __builtin_bit_cast(unsigned, f);
    unsigned r = (u + 0x7FFFu + ((u >> 16) & 1u)) >> 16;   // RNE
    unsigned short s = (unsigned short)r;
    return __builtin_bit_cast(bf16_t, s);
}

// ---------------------------------------------------------------------------
// fp32 -> bf16 elementwise (weight casts)
// ---------------------------------------------------------------------------
__global__ void f32_to_bf16_kernel(const float* __restrict__ in,
                                   bf16_t* __restrict__ out, int n) {
    int i = blockIdx.x * blockDim.x + threadIdx.x;
    if (i < n) out[i] = f2bf(in[i]);
}

// extract dt_low (cols 0..63 of x_dbl[M,96]) as bf16 [M,64]
__global__ void dtlow_convert_kernel(const float* __restrict__ xdbl,
                                     bf16_t* __restrict__ out) {
    int i = blockIdx.x * blockDim.x + threadIdx.x;
    if (i >= M_ * DTR_) return;
    int m = i >> 6, r = i & 63;
    out[i] = f2bf(xdbl[(size_t)m * 96 + r]);
}

// ---------------------------------------------------------------------------
// LayerNorm over D_MODEL=1024, one block (8 waves) per token row; bf16 out.
// ---------------------------------------------------------------------------
__global__ __launch_bounds__(256)
void layernorm_kernel(const float* __restrict__ x, const float* __restrict__ w,
                      const float* __restrict__ b, bf16_t* __restrict__ out) {
    __shared__ float red[2][8];
    int row = blockIdx.x;
    const float* xr = x + (size_t)row * DM_;
    float v[4], s = 0.f, ss = 0.f;
#pragma unroll
    for (int j = 0; j < 4; ++j) {
        v[j] = xr[threadIdx.x + 256 * j];
        s += v[j]; ss += v[j] * v[j];
    }
#pragma unroll
    for (int off = 16; off > 0; off >>= 1) {       // wave32 reduction
        s  += __shfl_down(s,  off, 32);
        ss += __shfl_down(ss, off, 32);
    }
    int wave = threadIdx.x >> 5, lane = threadIdx.x & 31;
    if (lane == 0) { red[0][wave] = s; red[1][wave] = ss; }
    __syncthreads();
    if (threadIdx.x == 0) {
        float t = 0.f, tt = 0.f;
        for (int i = 0; i < 8; ++i) { t += red[0][i]; tt += red[1][i]; }
        red[0][0] = t; red[1][0] = tt;
    }
    __syncthreads();
    float mean = red[0][0] * (1.f / DM_);
    float var  = red[1][0] * (1.f / DM_) - mean * mean;
    float rstd = rsqrtf(var + 1e-5f);
#pragma unroll
    for (int j = 0; j < 4; ++j) {
        int c = threadIdx.x + 256 * j;
        out[(size_t)row * DM_ + c] = f2bf((v[j] - mean) * rstd * w[c] + b[c]);
    }
}

// ---------------------------------------------------------------------------
// bf16 WMMA GEMM:  C[M,N] (+ residual) = A[M,K] * W[N,K]^T
//   - A row-major [M,K], W row-major [N,K] (PyTorch/JAX weight layout):
//     B fragment = one contiguous 32B run/lane, A fragment = two 16B runs.
//   - Wave tile 32x64: 2 A-frags x 4 B-frags -> 8 WMMAs per K-chunk.
//   - 8 waves (4 along M x 2 along N): block tile 128x128.  M % 128 == 0.
//   - OOB N columns: row pointer clamped (in-bounds load, garbage result),
//     discarded at store time -> no exec-mask work in the K-loop.
// ---------------------------------------------------------------------------
__global__ __launch_bounds__(256)
void gemm_bf16_kernel(const bf16_t* __restrict__ A, const bf16_t* __restrict__ W,
                      float* __restrict__ C, const float* __restrict__ residual,
                      int M, int N, int K) {
    const int tid  = threadIdx.x;
    const int wave = tid >> 5, lane = tid & 31;
    const int wm = wave >> 1, wn = wave & 1;
    const int m0    = blockIdx.y * 128 + wm * 32;
    const int nbase = blockIdx.x * 128 + wn * 64;

    const int am  = lane & 15;                 // A row within 16-tile
    const int akb = (lane < 16) ? 0 : 16;      // A K sub-base (contiguous, matches B)
    const int bn  = lane & 15;                 // B column within tile
    const int bkb = (lane < 16) ? 0 : 16;      // B K sub-base (contiguous, matches A)

    v8f acc[2][4] = {};
    const bf16_t* Arow0 = A + (size_t)(m0 + am) * K + akb;
    const bf16_t* Arow1 = A + (size_t)(m0 + 16 + am) * K + akb;
    const bf16_t* Wrow[4];
#pragma unroll
    for (int t = 0; t < 4; ++t) {
        int n = nbase + t * 16 + bn;
        int nc = (n < N) ? n : (N - 1);        // clamp: stay in-bounds
        Wrow[t] = W + (size_t)nc * K + bkb;
    }

    for (int k = 0; k < K; k += 32) {
        if (k + 32 < K) {                      // gfx1250 global_prefetch
            __builtin_prefetch(Arow0 + k + 32, 0, 3);
            __builtin_prefetch(Arow1 + k + 32, 0, 3);
        }
        v8bf a0l = *(const v8bf*)(Arow0 + k);        // K = kb..kb+7
        v8bf a0h = *(const v8bf*)(Arow0 + k + 8);    // K = kb+8..kb+15
        v8bf a1l = *(const v8bf*)(Arow1 + k);
        v8bf a1h = *(const v8bf*)(Arow1 + k + 8);
        v16bf a0, a1;
#pragma unroll
        for (int i = 0; i < 8; ++i) {
            a0[i] = a0l[i]; a0[i + 8] = a0h[i];
            a1[i] = a1l[i]; a1[i + 8] = a1h[i];
        }
#pragma unroll
        for (int t = 0; t < 4; ++t) {
            v16bf bf = *(const v16bf*)(Wrow[t] + k);  // 32B contiguous run
            acc[0][t] = __builtin_amdgcn_wmma_f32_16x16x32_bf16(
                false, a0, false, bf, (short)0, acc[0][t], false, false);
            acc[1][t] = __builtin_amdgcn_wmma_f32_16x16x32_bf16(
                false, a1, false, bf, (short)0, acc[1][t], false, false);
        }
    }

    const int chi = lane >> 4, cn = lane & 15;   // C/D layout (8 VGPRs)
#pragma unroll
    for (int r = 0; r < 2; ++r) {
#pragma unroll
        for (int t = 0; t < 4; ++t) {
            int n = nbase + t * 16 + cn;
            if (n >= N) continue;
#pragma unroll
            for (int v = 0; v < 8; ++v) {
                int m = m0 + r * 16 + v + 8 * chi;
                size_t idx = (size_t)m * N + n;
                float val = acc[r][t][v];
                if (residual) val += residual[idx];
                C[idx] = val;
            }
        }
    }
}

// ---------------------------------------------------------------------------
// Causal depthwise conv1d (k=4) + SiLU; reads xc = xz[:, :, 0:DI].
// ---------------------------------------------------------------------------
__global__ void conv_silu_kernel(const float* __restrict__ xz,
                                 const float* __restrict__ cw,
                                 const float* __restrict__ cb,
                                 float* __restrict__ u, bf16_t* __restrict__ ub) {
    int i = blockIdx.x * blockDim.x + threadIdx.x;
    if (i >= M_ * DI_) return;
    int d  = i % DI_;
    int bl = i / DI_;
    int l  = bl % L_;
    int b  = bl / L_;
    float acc = cb[d];
#pragma unroll
    for (int j = 0; j < DC_; ++j) {
        int lt = l - (DC_ - 1) + j;
        if (lt >= 0)
            acc += cw[d * DC_ + j] * xz[((size_t)(b * L_ + lt)) * (2 * DI_) + d];
    }
    float s = acc / (1.f + __expf(-acc));   // SiLU
    u[i]  = s;
    ub[i] = f2bf(s);
}

// softplus(dt + bias) in place
__global__ void softplus_kernel(float* __restrict__ dt,
                                const float* __restrict__ bias, int total) {
    int i = blockIdx.x * blockDim.x + threadIdx.x;
    if (i >= total) return;
    float x = dt[i] + bias[i % DI_];
    dt[i] = (x > 20.f) ? x : log1pf(__expf(x));
}

// ---------------------------------------------------------------------------
// Selective scan: thread = (b,d). h[16] in registers; per-32-step tiles of
// B_t/C_t broadcast via LDS (all threads in a block share b).
// ---------------------------------------------------------------------------
__global__ __launch_bounds__(256)
void scan_kernel(const float* __restrict__ dt, const float* __restrict__ xdbl,
                 const float* __restrict__ u, const float* __restrict__ A_log,
                 float* __restrict__ ys) {
    int gid = blockIdx.x * 256 + threadIdx.x;
    if (gid >= B_ * DI_) return;
    int b = gid / DI_;
    int d = gid % DI_;

    float Ar[DS_];
#pragma unroll
    for (int s = 0; s < DS_; ++s) Ar[s] = -__expf(A_log[d * DS_ + s]);
    float h[DS_];
#pragma unroll
    for (int s = 0; s < DS_; ++s) h[s] = 0.f;

    __shared__ float sB[32][DS_];
    __shared__ float sC[32][DS_];

    for (int l0 = 0; l0 < L_; l0 += 32) {
        __syncthreads();
        for (int t = threadIdx.x; t < 32 * DS_; t += 256) {
            int lt = t >> 4, s = t & 15;
            const float* row = xdbl + ((size_t)(b * L_) + l0 + lt) * 96;
            sB[lt][s] = row[DTR_ + s];         // B_t
            sC[lt][s] = row[DTR_ + DS_ + s];   // C_t
        }
        __syncthreads();
#pragma unroll 4
        for (int lt = 0; lt < 32; ++lt) {
            size_t idx = ((size_t)(b * L_ + l0 + lt)) * DI_ + d;
            float dtv = dt[idx];
            float uv  = u[idx];
            float du  = dtv * uv;
            float y   = 0.f;
#pragma unroll
            for (int s = 0; s < DS_; ++s) {
                float dA = __expf(dtv * Ar[s]);
                h[s] = fmaf(dA, h[s], du * sB[lt][s]);
                y    = fmaf(h[s], sC[lt][s], y);
            }
            ys[idx] = y;
        }
    }
}

// y = (ys + u*D) * silu(z); z = xz[:, :, DI + d]; out bf16 for out_proj GEMM.
__global__ void gate_kernel(const float* __restrict__ ys, const float* __restrict__ u,
                            const float* __restrict__ Dp, const float* __restrict__ xz,
                            bf16_t* __restrict__ yb) {
    int i = blockIdx.x * blockDim.x + threadIdx.x;
    if (i >= M_ * DI_) return;
    int d = i % DI_;
    size_t bl = (size_t)(i / DI_);
    float z = xz[bl * (2 * DI_) + DI_ + d];
    float y = (ys[i] + u[i] * Dp[d]) * (z / (1.f + __expf(-z)));
    yb[i] = f2bf(y);
}

// ---------------------------------------------------------------------------
extern "C" void kernel_launch(void* const* d_in, const int* in_sizes, int n_in,
                              void* d_out, int out_size, void* d_ws, size_t ws_size,
                              hipStream_t stream) {
    (void)in_sizes; (void)n_in; (void)out_size; (void)ws_size;
    const float* x          = (const float*)d_in[0];
    const float* norm_w     = (const float*)d_in[1];
    const float* norm_b     = (const float*)d_in[2];
    const float* in_proj_w  = (const float*)d_in[3];
    const float* conv_w     = (const float*)d_in[4];
    const float* conv_b     = (const float*)d_in[5];
    const float* x_proj_w   = (const float*)d_in[6];
    const float* dt_proj_w  = (const float*)d_in[7];
    const float* dt_proj_b  = (const float*)d_in[8];
    const float* A_log      = (const float*)d_in[9];
    const float* Dp         = (const float*)d_in[10];
    const float* out_proj_w = (const float*)d_in[11];
    float* out = (float*)d_out;

    char* ws = (char*)d_ws;
    size_t off = 0;
    auto alloc = [&](size_t bytes) -> char* {
        char* p = ws + off;
        off = (off + bytes + 255) & ~(size_t)255;
        return p;
    };
    bf16_t* xn_bf   = (bf16_t*)alloc((size_t)M_ * DM_ * 2);       // 8 MB
    bf16_t* w_in    = (bf16_t*)alloc((size_t)2 * DI_ * DM_ * 2);  // 8 MB
    bf16_t* w_xp    = (bf16_t*)alloc((size_t)(DTR_ + 2 * DS_) * DI_ * 2);
    bf16_t* w_dt    = (bf16_t*)alloc((size_t)DI_ * DTR_ * 2);
    bf16_t* w_op    = (bf16_t*)alloc((size_t)DM_ * DI_ * 2);      // 4 MB
    float*  xz      = (float*) alloc((size_t)M_ * 2 * DI_ * 4);   // 64 MB
    float*  u_f     = (float*) alloc((size_t)M_ * DI_ * 4);       // 32 MB
    bf16_t* u_bf    = (bf16_t*)alloc((size_t)M_ * DI_ * 2);       // 16 MB
    float*  x_dbl   = (float*) alloc((size_t)M_ * 96 * 4);
    bf16_t* dtlow   = (bf16_t*)alloc((size_t)M_ * DTR_ * 2);
    float*  dt_full = (float*) alloc((size_t)M_ * DI_ * 4);       // 32 MB
    float*  ys      = (float*) alloc((size_t)M_ * DI_ * 4);       // 32 MB
    bf16_t* y_bf    = (bf16_t*)alloc((size_t)M_ * DI_ * 2);       // 16 MB

    // 1) LayerNorm -> bf16 activations
    layernorm_kernel<<<M_, 256, 0, stream>>>(x, norm_w, norm_b, xn_bf);

    // 2) Weight casts fp32 -> bf16
    f32_to_bf16_kernel<<<(2 * DI_ * DM_ + 255) / 256, 256, 0, stream>>>(in_proj_w, w_in, 2 * DI_ * DM_);
    f32_to_bf16_kernel<<<((DTR_ + 2 * DS_) * DI_ + 255) / 256, 256, 0, stream>>>(x_proj_w, w_xp, (DTR_ + 2 * DS_) * DI_);
    f32_to_bf16_kernel<<<(DI_ * DTR_ + 255) / 256, 256, 0, stream>>>(dt_proj_w, w_dt, DI_ * DTR_);
    f32_to_bf16_kernel<<<(DM_ * DI_ + 255) / 256, 256, 0, stream>>>(out_proj_w, w_op, DM_ * DI_);

    // 3) in_proj: [4096,1024] x [4096,1024]^T -> xz [4096,4096]
    {
        dim3 g((2 * DI_ + 127) / 128, M_ / 128);
        gemm_bf16_kernel<<<g, 256, 0, stream>>>(xn_bf, w_in, xz, nullptr, M_, 2 * DI_, DM_);
    }

    // 4) depthwise conv + SiLU -> u (fp32 + bf16)
    conv_silu_kernel<<<(M_ * DI_ + 255) / 256, 256, 0, stream>>>(xz, conv_w, conv_b, u_f, u_bf);

    // 5) x_proj: [4096,2048] x [96,2048]^T -> x_dbl [4096,96]
    {
        dim3 g(1, M_ / 128);
        gemm_bf16_kernel<<<g, 256, 0, stream>>>(u_bf, w_xp, x_dbl, nullptr, M_, DTR_ + 2 * DS_, DI_);
    }
    dtlow_convert_kernel<<<(M_ * DTR_ + 255) / 256, 256, 0, stream>>>(x_dbl, dtlow);

    // 6) dt_proj: [4096,64] x [2048,64]^T -> dt_full [4096,2048], then softplus
    {
        dim3 g(DI_ / 128, M_ / 128);
        gemm_bf16_kernel<<<g, 256, 0, stream>>>(dtlow, w_dt, dt_full, nullptr, M_, DI_, DTR_);
    }
    softplus_kernel<<<(M_ * DI_ + 255) / 256, 256, 0, stream>>>(dt_full, dt_proj_b, M_ * DI_);

    // 7) selective scan (sequential in L, parallel over 4096 channels)
    scan_kernel<<<(B_ * DI_) / 256, 256, 0, stream>>>(dt_full, x_dbl, u_f, A_log, ys);

    // 8) gate: y = (ys + u*D) * silu(z) -> bf16
    gate_kernel<<<(M_ * DI_ + 255) / 256, 256, 0, stream>>>(ys, u_f, Dp, xz, y_bf);

    // 9) out_proj + residual: out = x + y_bf x [1024,2048]^T
    {
        dim3 g(DM_ / 128, M_ / 128);
        gemm_bf16_kernel<<<g, 256, 0, stream>>>(y_bf, w_op, out, x, M_, DM_, DI_);
    }
}
